// ATSPModel_55765855371380
// MI455X (gfx1250) — matrix operand order repaired and checked by reading, not compiled, not run on hardware
//
#include <hip/hip_runtime.h>
#include <hip/hip_bf16.h>

// ---------------- model dims ----------------
constexpr int Bb  = 8;
constexpr int Nn  = 256;   // nodes
constexpr int Dd  = 256;   // model dim
constexpr int Hh  = 16;    // heads
constexpr int DKk = 16;    // head dim
constexpr int MSs = 16;    // score-MLP hidden
constexpr int FFf = 512;   // feed-forward dim
constexpr int Mrows = Bb * Nn;           // 2048
constexpr float EPSv = 1e-5f;

typedef __attribute__((ext_vector_type(8)))  _Float16 v8h;
typedef __attribute__((ext_vector_type(16))) _Float16 v16h;
typedef __attribute__((ext_vector_type(8)))  float    v8f;

__device__ __forceinline__ v16h cat16(v8h lo, v8h hi) {
  return __builtin_shufflevector(lo, hi, 0,1,2,3,4,5,6,7,8,9,10,11,12,13,14,15);
}
__device__ __forceinline__ v8f wmma_f16(v16h a, v16h b, v8f c) {
  // D = A(16x32 f16) * B(32x16 f16) + C(16x16 f32)
  return __builtin_amdgcn_wmma_f32_16x16x32_f16(false, a, false, b, (short)0, c,
                                                false, false);
}

// ---------------- prep kernels ----------------
__global__ void k_f2h(const float* __restrict__ in, _Float16* __restrict__ out, int n) {
  int i = blockIdx.x * 256 + threadIdx.x;
  if (i < n) out[i] = (_Float16)in[i];
}

// in: R x C (row-major f32)  ->  out: C x R (row-major f16) == in^T
__global__ void k_transpose_f2h(const float* __restrict__ in, _Float16* __restrict__ out,
                                int R, int C) {
  int i = blockIdx.x * 256 + threadIdx.x;
  if (i < R * C) {
    int r = i / C, c = i % C;
    out[(size_t)c * R + r] = (_Float16)in[i];
  }
}

// ---------------- generic WMMA GEMM ----------------
// out[M,Nc] = X[M,K](f16) * WT[Nc,K]^T (f16) (+bias) (+relu) (+res) -> f32 and/or f16
// block = 128 threads (4 waves); each wave does 16 rows x 64 cols; grid (M/64, Nc/64)
template<bool RELU, bool VTSTORE>
__global__ void k_gemm(const _Float16* __restrict__ X, const _Float16* __restrict__ WT,
                       const float* __restrict__ bias, const float* __restrict__ res,
                       float* __restrict__ outF, _Float16* __restrict__ outH,
                       int Nc, int K) {
  const int tid  = threadIdx.x;
  const int wave = tid >> 5;
  const int l    = tid & 31, ln = l & 15, g = l >> 4;
  const int m0 = blockIdx.x * 64 + wave * 16;
  const int n0 = blockIdx.y * 64;

  v8f acc[4] = {};
  const _Float16* arow = X + (size_t)(m0 + ln) * K;

  for (int k0 = 0; k0 < K; k0 += 32) {
    __builtin_prefetch(arow + k0 + 64, 0, 1);
    v16h a = cat16(*(const v8h*)(arow + k0 + g * 8),
                   *(const v8h*)(arow + k0 + 16 + g * 8));
#pragma unroll
    for (int t = 0; t < 4; ++t) {
      const _Float16* brow = WT + (size_t)(n0 + t * 16 + ln) * K + k0 + g * 8;
      v16h bf = cat16(*(const v8h*)brow, *(const v8h*)(brow + 16));
      acc[t] = wmma_f16(a, bf, acc[t]);
    }
  }

#pragma unroll
  for (int t = 0; t < 4; ++t) {
#pragma unroll
    for (int r = 0; r < 8; ++r) {
      const int m = m0 + r + 8 * g;
      const int n = n0 + t * 16 + ln;
      float v = acc[t][r];
      if (bias) v += bias[n];
      if (RELU) v = fmaxf(v, 0.f);
      if (res)  v += res[(size_t)m * Nc + n];
      if (outF) outF[(size_t)m * Nc + n] = v;
      if (outH) {
        if (VTSTORE) {
          // m = b*256 + i ; n = h*16 + dk ; store as (b,h,dk,i)
          const int bi = m >> 8, i = m & 255, h = n >> 4, dk = n & 15;
          outH[(((size_t)(bi * Hh + h) * DKk + dk) * Nn) + i] = (_Float16)v;
        } else {
          outH[(size_t)m * Nc + n] = (_Float16)v;
        }
      }
    }
  }
}

// ---------------- fused QK^T (WMMA, K=16 zero-padded) + per-score MLP ----------------
// grid (jtiles=16, itiles=16, B*H=128); block = 32 (one wave per 16x16 score tile)
__global__ void k_attn_score(const _Float16* __restrict__ Q, const _Float16* __restrict__ Kt,
                             const float* __restrict__ cost, int costT,
                             const float* __restrict__ m1w, const float* __restrict__ m1b,
                             const float* __restrict__ m2w, const float* __restrict__ m2b,
                             _Float16* __restrict__ S) {
  const int l = threadIdx.x, ln = l & 15, g = l >> 4;
  const int bh = blockIdx.z, b = bh >> 4, h = bh & 15;
  const int i0 = blockIdx.y * 16, j0 = blockIdx.x * 16;

  const v8h zero = {};
  // A = q rows (K = dk 0..15 valid, 16..31 zero)
  const _Float16* qrow = Q + ((size_t)(b * Nn + i0 + ln) * Dd + h * DKk);
  v16h a = cat16(*(const v8h*)(qrow + g * 8), zero);
  // B = k^T columns: lanes carry K g*8..g*8+7 in low half, K 16..31 -> zero
  const _Float16* krow = Kt + ((size_t)(b * Nn + j0 + ln) * Dd + h * DKk);
  v8h blo = *(const v8h*)(krow + g * 8);
  v8h bhi = zero;
  v16h bf = cat16(blo, bhi);

  v8f c = {};
  c = wmma_f16(a, bf, c);

  // per-head score-MLP parameters (uniform within the wave)
  float w0[MSs], w1m[MSs], bm[MSs], w2m[MSs];
#pragma unroll
  for (int m = 0; m < MSs; ++m) {
    w0[m]  = m1w[h * 2 * MSs + m];
    w1m[m] = m1w[h * 2 * MSs + MSs + m];
    bm[m]  = m1b[h * MSs + m];
    w2m[m] = m2w[h * MSs + m];
  }
  const float b2v = m2b[h];

#pragma unroll
  for (int r = 0; r < 8; ++r) {
    const int i = i0 + r + 8 * g, j = j0 + ln;
    const float dotv = c[r] * 0.25f;  // 1/sqrt(DK)
    const float costv = costT ? cost[((size_t)b * Nn + j) * Nn + i]
                              : cost[((size_t)b * Nn + i) * Nn + j];
    float acc = b2v;
#pragma unroll
    for (int m = 0; m < MSs; ++m) {
      float t = fmaf(dotv, w0[m], fmaf(costv, w1m[m], bm[m]));
      t = fmaxf(t, 0.f);
      acc = fmaf(t, w2m[m], acc);
    }
    S[((size_t)bh * Nn + i) * Nn + j] = (_Float16)acc;
  }
}

// ---------------- softmax over last axis (in place, f16), one wave per row ----------------
__global__ void k_softmax(_Float16* __restrict__ S) {
  const int wave = threadIdx.x >> 5, l = threadIdx.x & 31;
  const size_t row = (size_t)blockIdx.x * 4 + wave;
  _Float16* p = S + row * Nn;
  float v[8];
  float mx = -1e30f;
#pragma unroll
  for (int t = 0; t < 8; ++t) { v[t] = (float)p[l + 32 * t]; mx = fmaxf(mx, v[t]); }
  for (int off = 16; off; off >>= 1) mx = fmaxf(mx, __shfl_xor(mx, off, 32));
  float s = 0.f;
#pragma unroll
  for (int t = 0; t < 8; ++t) { v[t] = __expf(v[t] - mx); s += v[t]; }
  for (int off = 16; off; off >>= 1) s += __shfl_xor(s, off, 32);
  const float inv = 1.0f / s;
#pragma unroll
  for (int t = 0; t < 8; ++t) p[l + 32 * t] = (_Float16)(v[t] * inv);
}

// ---------------- attention * V (WMMA), VT is (b,h,dk,j) ----------------
// grid (itiles=16, B*H=128); block = 32
__global__ void k_attn_v(const _Float16* __restrict__ W, const _Float16* __restrict__ VT,
                         _Float16* __restrict__ OC) {
  const int l = threadIdx.x, ln = l & 15, g = l >> 4;
  const int bh = blockIdx.y, b = bh >> 4, h = bh & 15;
  const int i0 = blockIdx.x * 16;
  const _Float16* arow = W + ((size_t)bh * Nn + i0 + ln) * Nn;
  const _Float16* brow = VT + ((size_t)bh * DKk + ln) * Nn;
  v8f c = {};
  for (int k0 = 0; k0 < Nn; k0 += 32) {
    v16h a  = cat16(*(const v8h*)(arow + k0 + g * 8),
                    *(const v8h*)(arow + k0 + 16 + g * 8));
    v16h bf = cat16(*(const v8h*)(brow + k0 + g * 8),
                    *(const v8h*)(brow + k0 + 16 + g * 8));
    c = wmma_f16(a, bf, c);
  }
#pragma unroll
  for (int r = 0; r < 8; ++r) {
    const int i = i0 + r + 8 * g;
    OC[((size_t)(b * Nn + i)) * Dd + h * DKk + ln] = (_Float16)c[r];
  }
}

// ---------------- InstanceNorm over node axis per (b,d) ----------------
// grid = B, block = D (256): one thread per channel, loop nodes
__global__ void k_inorm(const float* __restrict__ X, const float* __restrict__ scale,
                        const float* __restrict__ bias, float* __restrict__ outF,
                        _Float16* __restrict__ outH) {
  const int b = blockIdx.x, d = threadIdx.x;
  const float* p = X + (size_t)b * Nn * Dd + d;
  float s = 0.f, s2 = 0.f;
  for (int n = 0; n < Nn; ++n) { float v = p[(size_t)n * Dd]; s += v; s2 += v * v; }
  const float mean = s * (1.0f / Nn);
  const float var  = s2 * (1.0f / Nn) - mean * mean;   // biased var, like jnp.var
  const float a  = rsqrtf(var + EPSv) * scale[d];
  const float bc = bias[d] - mean * a;
  for (int n = 0; n < Nn; ++n) {
    const float v = fmaf(p[(size_t)n * Dd], a, bc);
    const size_t idx = (size_t)b * Nn * Dd + (size_t)n * Dd + d;
    outF[idx] = v;
    if (outH) outH[idx] = (_Float16)v;
  }
}

// ---------------- host orchestration ----------------
extern "C" void kernel_launch(void* const* d_in, const int* in_sizes, int n_in,
                              void* d_out, int out_size, void* d_ws, size_t ws_size,
                              hipStream_t stream) {
  (void)in_sizes; (void)n_in; (void)out_size; (void)ws_size;

  const float* row_emb = (const float*)d_in[0];
  const float* col_emb = (const float*)d_in[1];
  const float* cost    = (const float*)d_in[2];

  // per-side input index offsets (dict order)
  auto IN = [&](int side, int k) -> const float* {   // side: 0 = 'r', 1 = 'c'
    return (const float*)d_in[3 + side * 17 + k];
  };
  // k: 0 Wq,1 Wk,2 Wv,3 m1w,4 m1b,5 m2w,6 m2b,7 cw,8 cb,9 n1s,10 n1b,
  //    11 w1,12 b1,13 w2,14 b2,15 n2s,16 n2b

  // ---- workspace carve-up ----
  char* p = (char*)d_ws;
  auto take = [&](size_t bytes) { char* q = p; p += (bytes + 255) & ~(size_t)255; return q; };
  const size_t EMB16 = (size_t)Mrows * Dd * sizeof(_Float16);         // 1 MB
  const size_t W256  = (size_t)Dd * Dd * sizeof(_Float16);            // 128 KB
  const size_t WFF   = (size_t)Dd * FFf * sizeof(_Float16);           // 256 KB

  _Float16* Xr16 = (_Float16*)take(EMB16);
  _Float16* Xc16 = (_Float16*)take(EMB16);
  _Float16* WT[2][6];   // WqT, WkT, WvT, cwT, w1T, w2T  (all stored as [out][in]^T = [n][k])
  for (int s = 0; s < 2; ++s) {
    WT[s][0] = (_Float16*)take(W256);
    WT[s][1] = (_Float16*)take(W256);
    WT[s][2] = (_Float16*)take(W256);
    WT[s][3] = (_Float16*)take(W256);
    WT[s][4] = (_Float16*)take(WFF);   // w1T: (FF, D)
    WT[s][5] = (_Float16*)take(WFF);   // w2T: (D, FF)
  }
  _Float16* Q16  = (_Float16*)take(EMB16);
  _Float16* K16  = (_Float16*)take(EMB16);
  _Float16* VT16 = (_Float16*)take(EMB16);
  _Float16* S16  = (_Float16*)take((size_t)Bb * Hh * Nn * Nn * sizeof(_Float16)); // 16 MB
  _Float16* OC16 = (_Float16*)take(EMB16);
  float*    X1   = (float*)take((size_t)Mrows * Dd * sizeof(float));
  float*    OUT1 = (float*)take((size_t)Mrows * Dd * sizeof(float));
  _Float16* O116 = (_Float16*)take(EMB16);
  _Float16* HH16 = (_Float16*)take((size_t)Mrows * FFf * sizeof(_Float16));       // 2 MB
  float*    X2   = (float*)take((size_t)Mrows * Dd * sizeof(float));

  // ---- prep: f16 embeddings + transposed f16 weights ----
  {
    const int n = Mrows * Dd;
    k_f2h<<<(n + 255) / 256, 256, 0, stream>>>(row_emb, Xr16, n);
    k_f2h<<<(n + 255) / 256, 256, 0, stream>>>(col_emb, Xc16, n);
  }
  for (int s = 0; s < 2; ++s) {
    k_transpose_f2h<<<(Dd * Dd + 255) / 256, 256, 0, stream>>>(IN(s, 0), WT[s][0], Dd, Dd);
    k_transpose_f2h<<<(Dd * Dd + 255) / 256, 256, 0, stream>>>(IN(s, 1), WT[s][1], Dd, Dd);
    k_transpose_f2h<<<(Dd * Dd + 255) / 256, 256, 0, stream>>>(IN(s, 2), WT[s][2], Dd, Dd);
    k_transpose_f2h<<<(Dd * Dd + 255) / 256, 256, 0, stream>>>(IN(s, 7), WT[s][3], Dd, Dd);
    k_transpose_f2h<<<(Dd * FFf + 255) / 256, 256, 0, stream>>>(IN(s, 11), WT[s][4], Dd, FFf);
    k_transpose_f2h<<<(Dd * FFf + 255) / 256, 256, 0, stream>>>(IN(s, 13), WT[s][5], FFf, Dd);
  }

  // ---- one transformer block ----
  auto run_block = [&](int s, const _Float16* Xq, const _Float16* Xkv,
                       const float* res_emb, int costT, float* out_ptr) {
    const dim3 blk128(128), blk32(32), blk256(256);
    const dim3 gD(Mrows / 64, Dd / 64);    // (32,4)
    const dim3 gFF(Mrows / 64, FFf / 64);  // (32,8)

    // QKV projections
    k_gemm<false, false><<<gD, blk128, 0, stream>>>(Xq,  WT[s][0], nullptr, nullptr,
                                                    nullptr, Q16, Dd, Dd);
    k_gemm<false, false><<<gD, blk128, 0, stream>>>(Xkv, WT[s][1], nullptr, nullptr,
                                                    nullptr, K16, Dd, Dd);
    k_gemm<false, true ><<<gD, blk128, 0, stream>>>(Xkv, WT[s][2], nullptr, nullptr,
                                                    nullptr, VT16, Dd, Dd);
    // scores = MLP(qk/sqrt(dk), cost)
    k_attn_score<<<dim3(Nn / 16, Nn / 16, Bb * Hh), blk32, 0, stream>>>(
        Q16, K16, cost, costT, IN(s, 3), IN(s, 4), IN(s, 5), IN(s, 6), S16);
    // softmax rows
    k_softmax<<<(Bb * Hh * Nn) / 4, blk128, 0, stream>>>(S16);
    // out = softmax @ V  -> concat heads
    k_attn_v<<<dim3(Nn / 16, Bb * Hh), blk32, 0, stream>>>(S16, VT16, OC16);
    // mh = OC @ cw + cb ; x1 = emb + mh
    k_gemm<false, false><<<gD, blk128, 0, stream>>>(OC16, WT[s][3], IN(s, 8), res_emb,
                                                    X1, nullptr, Dd, Dd);
    // out1 = inorm(x1)
    k_inorm<<<Bb, blk256, 0, stream>>>(X1, IN(s, 9), IN(s, 10), OUT1, O116);
    // ff hidden = relu(out1 @ w1 + b1)
    k_gemm<true, false><<<gFF, blk128, 0, stream>>>(O116, WT[s][4], IN(s, 12), nullptr,
                                                    nullptr, HH16, FFf, Dd);
    // x2 = out1 + hidden @ w2 + b2
    k_gemm<false, false><<<gD, blk128, 0, stream>>>(HH16, WT[s][5], IN(s, 14), OUT1,
                                                    X2, nullptr, Dd, FFf);
    // final norm -> output
    k_inorm<<<Bb, blk256, 0, stream>>>(X2, IN(s, 15), IN(s, 16), out_ptr, nullptr);
  };

  float* out = (float*)d_out;
  // row block: q from row_emb, k/v from col_emb, cost as-is
  run_block(0, Xr16, Xc16, row_emb, 0, out);
  // col block: q from col_emb, k/v from row_emb, cost transposed
  run_block(1, Xc16, Xr16, col_emb, 1, out + (size_t)Mrows * Dd);
}
